// HybridSignatureAttentionHedger_48473000902845
// MI455X (gfx1250) — matrix operand order repaired and checked by reading, not compiled, not run on hardware
//
#include <hip/hip_runtime.h>
#include <hip/hip_bf16.h>

// ---------------------------------------------------------------------------
// Fused HybridSignatureAttentionHedger for gfx1250 (MI455X).
// Compute-bound (~5.6e10 FLOP vs ~12 MB HBM traffic at 23.3 TB/s) -> every
// linear layer runs on v_wmma_f32_16x16x32_f16 with batch rows as M and
// f16 weights pre-packed in LDS in B-fragment layout. One block = 64 batch
// rows through the whole T=30 scan; 8 wave32 per block, 256 blocks.
// ---------------------------------------------------------------------------

typedef __attribute__((ext_vector_type(16))) _Float16 v16h;
typedef __attribute__((ext_vector_type(8)))  _Float16 h8v;
typedef __attribute__((ext_vector_type(8)))  float    v8f;

#define DEVI __device__ __forceinline__

constexpr int BTOT = 16384;
constexpr int TT   = 30;
constexpr int CF   = 5;
constexpr int WINW = 5;
constexpr int DD   = 6;       // C+1
constexpr int MB   = 64;      // batch rows per block
constexpr int NTHR = 256;     // 8 wave32

// packed-weight sub offsets (halves); fragment = 32 lanes * 16 halves = 512
constexpr int W_LVL0 = 0;          // 1 kb x 4 nt
constexpr int W_LVL1 = 2048;       // 2 x 4
constexpr int W_LVL2 = 6144;       // 7 x 4
constexpr int W_SE1  = 20480;      // 1 x 4
constexpr int W_SE2  = 22528;      // 2 x 4
constexpr int W_Q    = 26624;      // 2 x 4
constexpr int W_K    = 30720;      // 2 x 4
constexpr int W_V    = 34816;      // 2 x 4
constexpr int W_O    = 38912;      // 2 x 4
constexpr int W_DH1  = 43008;      // 3 x 2
constexpr int W_DH2  = 46080;      // 1 x 2
constexpr int W_DH3  = 47104;      // 1 x 1

// LDS param-cache offsets (floats)
constexpr int PRM_LVLB0 = 0;      // b/g/beta per level: i*192 + {0,64,128}
constexpr int PRM_SE1B  = 576;
constexpr int PRM_SE2B  = 640;
constexpr int PRM_QB    = 704;
constexpr int PRM_KB    = 768;
constexpr int PRM_VB    = 832;
constexpr int PRM_OB    = 896;
constexpr int PRM_LNG   = 960;
constexpr int PRM_LNB   = 1024;
constexpr int PRM_DH1B  = 1088;
constexpr int PRM_DH2B  = 1120;
constexpr int PRM_DH3B  = 1152;

// dynamic-LDS byte offsets (all 16B aligned)
constexpr int OFF_WPACK  = 0;        // 95232 B f16 packed weights
constexpr int OFF_ASTAGE = 95232;    // [64][320] f16: s1@0, s2@32, s3@96 / act
constexpr int OFF_STG64  = 136192;   // [64][64]  f16 (se1 relu stage)
constexpr int OFF_DST1   = 144384;   // [64][32]  f16
constexpr int OFF_DST2   = 148480;   // [64][32]  f16
constexpr int OFF_TOKB   = 152576;   // [3][64][64] f16 tokens
constexpr int OFF_KBUF   = 177152;   // [64][192] f16 K
constexpr int OFF_VBUF   = 201728;   // [64][192] f16 V
constexpr int OFF_EBUF   = 226304;   // [3][64][64] f32 GEMM outs
constexpr int OFF_HBUF   = 275456;   // [64][64] f32 h (residual)
constexpr int OFF_PATH   = 291840;   // [64][5][6] f32 path windows
constexpr int OFF_DELTA  = 299520;   // [64] f32 running delta
constexpr int OFF_PRM    = 299776;   // 4624 B f32 bias/LN cache
constexpr int SMEM_BYTES = 304400;   // ~297 KB < 320 KB/WGP

struct KParams { const float* p[33]; };

// ---- WMMA fragment helpers (layouts per CDNA5 ISA 7.12.2) ------------------

// A (16x32 f16): lane l<16 holds row M=l, K 0..7 & 16..23; lane l+16 holds
// row M=l, K 8..15 & 24..31 -> two contiguous 16B LDS reads per lane.
DEVI v16h load_a(const _Float16* A, int stride, int row0, int k0) {
  int lane = threadIdx.x & 31;
  int l  = lane & 15;
  int hh = lane >> 4;
  const _Float16* base = A + (row0 + l) * stride + k0 + hh * 8;
  h8v lo = *(const h8v*)(base);
  h8v hi = *(const h8v*)(base + 16);
  v16h r;
#pragma unroll
  for (int e = 0; e < 8; ++e) { r[e] = lo[e]; r[8 + e] = hi[e]; }
  return r;
}

// B fragments pre-packed contiguously: frag*512 + lane*16 halves.
DEVI v16h load_b(const _Float16* Bp, int frag) {
  const _Float16* q = Bp + frag * 512 + (threadIdx.x & 31) * 16;
  h8v a0 = *(const h8v*)(q);
  h8v a1 = *(const h8v*)(q + 8);
  v16h r;
#pragma unroll
  for (int e = 0; e < 8; ++e) { r[e] = a0[e]; r[8 + e] = a1[e]; }
  return r;
}

DEVI void wave_gemm(const _Float16* A, int astride, int row0,
                    const _Float16* Bp, int kblocks, int ntot,
                    int nt0, int nt1, v8f& c0, v8f& c1, bool do1) {
  for (int kb = 0; kb < kblocks; ++kb) {
    v16h a  = load_a(A, astride, row0, kb * 32);
    v16h b0 = load_b(Bp, kb * ntot + nt0);
    c0 = __builtin_amdgcn_wmma_f32_16x16x32_f16(false, a, false, b0,
                                                (short)0, c0, false, false);
    if (do1) {
      v16h b1 = load_b(Bp, kb * ntot + nt1);
      c1 = __builtin_amdgcn_wmma_f32_16x16x32_f16(false, a, false, b1,
                                                  (short)0, c1, false, false);
    }
  }
}

// C/D 16x16 f32: lane l<16 VGPR r -> (M=row0+r, N=col0+l); lanes 16.. -> M+8.
DEVI void store_tile_f32(float* E, int stride, int row0, int col0,
                         const float* bias16, v8f c) {
  int lane = threadIdx.x & 31;
  int l  = lane & 15;
  int m0 = row0 + ((lane >> 4) << 3);
  float bb = bias16[l];
#pragma unroll
  for (int r = 0; r < 8; ++r)
    E[(m0 + r) * stride + col0 + l] = c[r] + bb;
}

DEVI void store_tile_f16(_Float16* E, int stride, int row0, int col0,
                         const float* bias16, bool dorelu, v8f c) {
  int lane = threadIdx.x & 31;
  int l  = lane & 15;
  int m0 = row0 + ((lane >> 4) << 3);
  float bb = bias16[l];
#pragma unroll
  for (int r = 0; r < 8; ++r) {
    float v = c[r] + bb;
    if (dorelu) v = fmaxf(v, 0.f);
    E[(m0 + r) * stride + col0 + l] = (_Float16)v;
  }
}

// Pack KxN row-major f32 weight into WMMA B-fragment f16 layout.
DEVI void pack_w(const float* W, int K, int N, int KB, int NTOT, _Float16* dst) {
  int total = KB * NTOT * 512;
  for (int idx = threadIdx.x; idx < total; idx += NTHR) {
    int f = idx >> 9, rem = idx & 511;
    int lane = rem >> 4, e = rem & 15;
    int kb = f / NTOT, nt = f % NTOT;
    int kr = kb * 32 + ((lane >> 4) << 3) + (e & 7) + ((e >> 3) << 4);
    int cn = nt * 16 + (lane & 15);
    float v = (kr < K && cn < N) ? W[kr * N + cn] : 0.f;
    dst[idx] = (_Float16)v;
  }
}

DEVI void cpv(float* dst, const float* src, int n) {
  for (int i = threadIdx.x; i < n; i += NTHR) dst[i] = src[i];
}

// sum across the 4 lanes of a quad (wave32 cross-lane)
DEVI float quad_sum(float v) {
  v += __shfl_xor(v, 1, 32);
  v += __shfl_xor(v, 2, 32);
  return v;
}

// ---------------------------------------------------------------------------

__global__ __launch_bounds__(NTHR, 1)
void hedger_fused(KParams kp, float* __restrict__ out) {
  extern __shared__ char smem[];
  _Float16* wpack  = (_Float16*)(smem + OFF_WPACK);
  _Float16* astage = (_Float16*)(smem + OFF_ASTAGE);   // stride 320
  _Float16* stg64  = (_Float16*)(smem + OFF_STG64);
  _Float16* dst1   = (_Float16*)(smem + OFF_DST1);
  _Float16* dst2   = (_Float16*)(smem + OFF_DST2);
  _Float16* tokb   = (_Float16*)(smem + OFF_TOKB);
  _Float16* kbuf   = (_Float16*)(smem + OFF_KBUF);
  _Float16* vbuf   = (_Float16*)(smem + OFF_VBUF);
  float*    ebuf0  = (float*)(smem + OFF_EBUF);
  float*    ebuf1  = ebuf0 + MB * 64;
  float*    ebuf2  = ebuf1 + MB * 64;
  float*    hbuf   = (float*)(smem + OFF_HBUF);
  float*    pathb  = (float*)(smem + OFF_PATH);
  float*    deltab = (float*)(smem + OFF_DELTA);
  float*    prm    = (float*)(smem + OFF_PRM);

  const float* feat = kp.p[0];
  const int tid  = threadIdx.x;
  const int wave = tid >> 5;
  const int p    = wave & 1;
  const int row0 = (wave >> 1) << 4;
  const int nt0  = p * 2, nt1 = p * 2 + 1;
  const int b0   = blockIdx.x * MB;
  const int qr4  = tid >> 2, qq = tid & 3;   // (row, quarter/head) mapping

  // ---- one-time: pack weights + cache params in LDS ----------------------
  pack_w(kp.p[1],  6,  64, 1, 4, wpack + W_LVL0);
  pack_w(kp.p[5],  36, 64, 2, 4, wpack + W_LVL1);
  pack_w(kp.p[9],  216,64, 7, 4, wpack + W_LVL2);
  pack_w(kp.p[13], 6,  64, 1, 4, wpack + W_SE1);
  pack_w(kp.p[15], 64, 64, 2, 4, wpack + W_SE2);
  pack_w(kp.p[17], 64, 64, 2, 4, wpack + W_Q);
  pack_w(kp.p[19], 64, 64, 2, 4, wpack + W_K);
  pack_w(kp.p[21], 64, 64, 2, 4, wpack + W_V);
  pack_w(kp.p[23], 64, 64, 2, 4, wpack + W_O);
  pack_w(kp.p[27], 65, 32, 3, 2, wpack + W_DH1);
  pack_w(kp.p[29], 32, 32, 1, 2, wpack + W_DH2);
  pack_w(kp.p[31], 32, 1,  1, 1, wpack + W_DH3);
  cpv(prm + PRM_LVLB0 + 0,   kp.p[2], 64);
  cpv(prm + PRM_LVLB0 + 64,  kp.p[3], 64);
  cpv(prm + PRM_LVLB0 + 128, kp.p[4], 64);
  cpv(prm + PRM_LVLB0 + 192, kp.p[6], 64);
  cpv(prm + PRM_LVLB0 + 256, kp.p[7], 64);
  cpv(prm + PRM_LVLB0 + 320, kp.p[8], 64);
  cpv(prm + PRM_LVLB0 + 384, kp.p[10], 64);
  cpv(prm + PRM_LVLB0 + 448, kp.p[11], 64);
  cpv(prm + PRM_LVLB0 + 512, kp.p[12], 64);
  cpv(prm + PRM_SE1B, kp.p[14], 64);
  cpv(prm + PRM_SE2B, kp.p[16], 64);
  cpv(prm + PRM_QB,   kp.p[18], 64);
  cpv(prm + PRM_KB,   kp.p[20], 64);
  cpv(prm + PRM_VB,   kp.p[22], 64);
  cpv(prm + PRM_OB,   kp.p[24], 64);
  cpv(prm + PRM_LNG,  kp.p[25], 64);
  cpv(prm + PRM_LNB,  kp.p[26], 64);
  cpv(prm + PRM_DH1B, kp.p[28], 32);
  cpv(prm + PRM_DH2B, kp.p[30], 32);
  cpv(prm + PRM_DH3B, kp.p[32], 1);
  if (tid < MB) deltab[tid] = 0.f;
  __syncthreads();

  for (int t = 0; t < TT; ++t) {
    if (t + 1 < TT && tid < MB)
      __builtin_prefetch(&feat[((size_t)(b0 + tid) * TT + t + 1) * CF], 0, 1);

    // 1 -- path windows: path[r][w][i], i=0 is time coord -------------------
    for (int idx = tid; idx < MB * WINW * DD; idx += NTHR) {
      int r = idx / (WINW * DD), rem = idx % (WINW * DD);
      int w = rem / DD, i = rem % DD;
      float v;
      if (i == 0) v = 0.25f * (float)w;
      else {
        int tc = t - (WINW - 1) + w; if (tc < 0) tc = 0;
        v = feat[((size_t)(b0 + r) * TT + tc) * CF + (i - 1)];
      }
      pathb[idx] = v;
    }
    __syncthreads();

    // 2 -- fill all signature A operands: s1@0(32), s2@32(64), s3@96(224) ---
    for (int idx = tid; idx < MB * 320; idx += NTHR) {
      int r = idx / 320, c = idx % 320;
      const float* pr = pathb + r * 30;
      float v = 0.f;
      if (c < 32) {
        if (c < DD) v = pr[4 * DD + c] - pr[c];
      } else if (c < 96) {
        int cc = c - 32;
        if (cc < 36) {
          int i = cc / 6, j = cc % 6;
#pragma unroll
          for (int w = 0; w < 4; ++w)
            v += (pr[(w + 1) * 6 + i] - pr[w * 6 + i]) * pr[w * 6 + j];
        }
      } else {
        int cc = c - 96;
        if (cc < 216) {
          int i = cc / 36, j = (cc / 6) % 6, k = cc % 6;
#pragma unroll
          for (int w = 0; w < 4; ++w)
            v += (pr[(w + 1) * 6 + i] - pr[w * 6 + i]) * pr[w * 6 + j] * pr[w * 6 + k];
          v *= 0.25f; // / (WIN-1)
        }
      }
      astage[r * 320 + c] = (_Float16)v;
    }
    __syncthreads();

    // 3 -- three level GEMMs back-to-back (10 k-blocks of wmma per wave) ----
    {
      v8f a0 = {}, a1 = {};
      wave_gemm(astage + 0,  320, row0, wpack + W_LVL0, 1, 4, nt0, nt1, a0, a1, true);
      store_tile_f32(ebuf0, 64, row0, nt0 * 16, prm + 0 + nt0 * 16, a0);
      store_tile_f32(ebuf0, 64, row0, nt1 * 16, prm + 0 + nt1 * 16, a1);
      v8f b0 = {}, b1 = {};
      wave_gemm(astage + 32, 320, row0, wpack + W_LVL1, 2, 4, nt0, nt1, b0, b1, true);
      store_tile_f32(ebuf1, 64, row0, nt0 * 16, prm + 192 + nt0 * 16, b0);
      store_tile_f32(ebuf1, 64, row0, nt1 * 16, prm + 192 + nt1 * 16, b1);
      v8f c0 = {}, c1 = {};
      wave_gemm(astage + 96, 320, row0, wpack + W_LVL2, 7, 4, nt0, nt1, c0, c1, true);
      store_tile_f32(ebuf2, 64, row0, nt0 * 16, prm + 384 + nt0 * 16, c0);
      store_tile_f32(ebuf2, 64, row0, nt1 * 16, prm + 384 + nt1 * 16, c1);
    }
    __syncthreads();

    // 4 -- LN+relu -> toks (quad-parallel, wave32 shuffles); state fill ----
    {
      const float* eb[3] = {ebuf0, ebuf1, ebuf2};
#pragma unroll
      for (int lvl = 0; lvl < 3; ++lvl) {
        const float* er = eb[lvl] + qr4 * 64 + qq * 16;
        float s = 0.f;
#pragma unroll
        for (int c = 0; c < 16; ++c) s += er[c];
        float mu = quad_sum(s) * (1.f / 64.f);
        float vs = 0.f;
#pragma unroll
        for (int c = 0; c < 16; ++c) { float d = er[c] - mu; vs += d * d; }
        float inv = rsqrtf(quad_sum(vs) * (1.f / 64.f) + 1e-5f);
        const float* g  = prm + lvl * 192 + 64  + qq * 16;
        const float* be = prm + lvl * 192 + 128 + qq * 16;
        _Float16* dr = tokb + lvl * MB * 64 + qr4 * 64 + qq * 16;
#pragma unroll
        for (int c = 0; c < 16; ++c)
          dr[c] = (_Float16)fmaxf(0.f, (er[c] - mu) * inv * g[c] + be[c]);
      }
    }
    for (int idx = tid; idx < MB * 32; idx += NTHR) {
      int r = idx >> 5, c = idx & 31;
      float v = 0.f;
      if (c < CF)       v = feat[((size_t)(b0 + r) * TT + t) * CF + c];
      else if (c == CF) v = deltab[r];
      astage[r * 320 + c] = (_Float16)v;
    }
    __syncthreads();

    // 5 -- K,V per token + SE1 ---------------------------------------------
    for (int j = 0; j < 3; ++j) {
      v8f c0 = {}, c1 = {};
      wave_gemm(tokb + j * MB * 64, 64, row0, wpack + W_K, 2, 4, nt0, nt1, c0, c1, true);
      store_tile_f16(kbuf, 192, row0, j * 64 + nt0 * 16, prm + PRM_KB + nt0 * 16, false, c0);
      store_tile_f16(kbuf, 192, row0, j * 64 + nt1 * 16, prm + PRM_KB + nt1 * 16, false, c1);
      v8f d0 = {}, d1 = {};
      wave_gemm(tokb + j * MB * 64, 64, row0, wpack + W_V, 2, 4, nt0, nt1, d0, d1, true);
      store_tile_f16(vbuf, 192, row0, j * 64 + nt0 * 16, prm + PRM_VB + nt0 * 16, false, d0);
      store_tile_f16(vbuf, 192, row0, j * 64 + nt1 * 16, prm + PRM_VB + nt1 * 16, false, d1);
    }
    {
      v8f c0 = {}, c1 = {};
      wave_gemm(astage, 320, row0, wpack + W_SE1, 1, 4, nt0, nt1, c0, c1, true);
      store_tile_f16(stg64, 64, row0, nt0 * 16, prm + PRM_SE1B + nt0 * 16, true, c0);
      store_tile_f16(stg64, 64, row0, nt1 * 16, prm + PRM_SE1B + nt1 * 16, true, c1);
    }
    __syncthreads();

    // 6 -- SE2: h -> hbuf f32 (residual) + astage f16 (A for q) ------------
    {
      v8f c0 = {}, c1 = {};
      wave_gemm(stg64, 64, row0, wpack + W_SE2, 2, 4, nt0, nt1, c0, c1, true);
      store_tile_f32(hbuf, 64, row0, nt0 * 16, prm + PRM_SE2B + nt0 * 16, c0);
      store_tile_f32(hbuf, 64, row0, nt1 * 16, prm + PRM_SE2B + nt1 * 16, c1);
      store_tile_f16(astage, 320, row0, nt0 * 16, prm + PRM_SE2B + nt0 * 16, false, c0);
      store_tile_f16(astage, 320, row0, nt1 * 16, prm + PRM_SE2B + nt1 * 16, false, c1);
    }
    __syncthreads();

    // 7 -- Q ----------------------------------------------------------------
    {
      v8f c0 = {}, c1 = {};
      wave_gemm(astage, 320, row0, wpack + W_Q, 2, 4, nt0, nt1, c0, c1, true);
      store_tile_f32(ebuf0, 64, row0, nt0 * 16, prm + PRM_QB + nt0 * 16, c0);
      store_tile_f32(ebuf0, 64, row0, nt1 * 16, prm + PRM_QB + nt1 * 16, c1);
    }
    __syncthreads();

    // 8 -- attention: one thread per (row, head), all 256 lanes busy -------
    {
      int r = qr4, h = qq;
      const float*    qh = ebuf0 + r * 64 + h * 16;
      const _Float16* kr = kbuf + r * 192 + h * 16;
      const _Float16* vr = vbuf + r * 192 + h * 16;
      float s[3];
#pragma unroll
      for (int j = 0; j < 3; ++j) {
        float a = 0.f;
#pragma unroll
        for (int d = 0; d < 16; ++d) a += qh[d] * (float)kr[j * 64 + d];
        s[j] = a * 0.25f; // 1/sqrt(HD)
      }
      float m = fmaxf(s[0], fmaxf(s[1], s[2]));
      float e0 = __expf(s[0] - m), e1 = __expf(s[1] - m), e2 = __expf(s[2] - m);
      float rs = 1.f / (e0 + e1 + e2);
      _Float16* orow = astage + r * 320 + h * 16;
#pragma unroll
      for (int d = 0; d < 16; ++d) {
        float ov = e0 * (float)vr[0 * 64 + d] + e1 * (float)vr[1 * 64 + d]
                 + e2 * (float)vr[2 * 64 + d];
        orow[d] = (_Float16)(ov * rs);
      }
    }
    __syncthreads();

    // 9 -- O projection -----------------------------------------------------
    {
      v8f c0 = {}, c1 = {};
      wave_gemm(astage, 320, row0, wpack + W_O, 2, 4, nt0, nt1, c0, c1, true);
      store_tile_f32(ebuf0, 64, row0, nt0 * 16, prm + PRM_OB + nt0 * 16, c0);
      store_tile_f32(ebuf0, 64, row0, nt1 * 16, prm + PRM_OB + nt1 * 16, c1);
    }
    __syncthreads();

    // 10 -- residual + LN (quad-parallel); x = [att, prev_delta, 0..] ------
    {
      int r = qr4;
      const float* er = ebuf0 + r * 64 + qq * 16;
      const float* hr = hbuf  + r * 64 + qq * 16;
      float s = 0.f;
#pragma unroll
      for (int c = 0; c < 16; ++c) s += er[c] + hr[c];
      float mu = quad_sum(s) * (1.f / 64.f);
      float vs = 0.f;
#pragma unroll
      for (int c = 0; c < 16; ++c) { float d = er[c] + hr[c] - mu; vs += d * d; }
      float inv = rsqrtf(quad_sum(vs) * (1.f / 64.f) + 1e-5f);
      const float* g  = prm + PRM_LNG + qq * 16;
      const float* be = prm + PRM_LNB + qq * 16;
      _Float16* xrow = astage + r * 320;
#pragma unroll
      for (int c = 0; c < 16; ++c)
        xrow[qq * 16 + c] = (_Float16)((er[c] + hr[c] - mu) * inv * g[c] + be[c]);
      if (qq == 0) xrow[64] = (_Float16)deltab[r];
      for (int c = 65 + qq; c < 96; c += 4) xrow[c] = (_Float16)0.f;
    }
    __syncthreads();

    // 11 -- delta head 65->32->32->1; same-wave LDS deps, no barriers ------
    if (p == 0) {
      v8f c0 = {}, c1 = {};
      wave_gemm(astage, 320, row0, wpack + W_DH1, 3, 2, 0, 1, c0, c1, true);
      store_tile_f16(dst1, 32, row0, 0,  prm + PRM_DH1B + 0,  true, c0);
      store_tile_f16(dst1, 32, row0, 16, prm + PRM_DH1B + 16, true, c1);
      v8f d0 = {}, d1 = {};
      wave_gemm(dst1, 32, row0, wpack + W_DH2, 1, 2, 0, 1, d0, d1, true);
      store_tile_f16(dst2, 32, row0, 0,  prm + PRM_DH2B + 0,  true, d0);
      store_tile_f16(dst2, 32, row0, 16, prm + PRM_DH2B + 16, true, d1);
      v8f g0 = {}, gD = {};
      wave_gemm(dst2, 32, row0, wpack + W_DH3, 1, 1, 0, 0, g0, gD, false);
      int lane = tid & 31, l = lane & 15, m0 = row0 + ((lane >> 4) << 3);
      if (l == 0) {
        float b3 = prm[PRM_DH3B];
#pragma unroll
        for (int r = 0; r < 8; ++r) ebuf0[(m0 + r) * 64] = g0[r] + b3;
      }
    }
    __syncthreads();

    // 12 -- scan update + output -------------------------------------------
    if (tid < MB) {
      float raw = ebuf0[tid * 64];
      float nd = deltab[tid] + 0.3f * tanhf(raw);
      nd = fminf(1.5f, fmaxf(-1.5f, nd));
      deltab[tid] = nd;
      out[(size_t)(b0 + tid) * TT + t] = nd;
    }
    __syncthreads();
  }
}

// ---------------------------------------------------------------------------

extern "C" void kernel_launch(void* const* d_in, const int* in_sizes, int n_in,
                              void* d_out, int out_size, void* d_ws, size_t ws_size,
                              hipStream_t stream) {
  (void)in_sizes; (void)out_size; (void)d_ws; (void)ws_size;
  KParams kp;
  for (int i = 0; i < 33; ++i)
    kp.p[i] = (i < n_in) ? (const float*)d_in[i] : nullptr;
  dim3 grid(BTOT / MB);   // 256 blocks, one 64-row batch slab each
  dim3 block(NTHR);       // 8 wave32
  hedger_fused<<<grid, block, SMEM_BYTES, stream>>>(kp, (float*)d_out);
}
